// Conv2d_44143673868891
// MI455X (gfx1250) — matrix operand order repaired and checked by reading, not compiled, not run on hardware
//
#include <hip/hip_runtime.h>
#include <hip/hip_bf16.h>

// CDNA5 (gfx1250) implicit-GEMM conv 3x3 s1 p1, bf16x3 split precision.
// Pass 1: weights -> bf16 hi/lo, K reordered to (pos, ci);
//         x -> bf16 hi/lo, zero-padded to 58x58, NHWC (ci contiguous).
// Pass 2: LDS-free, barrier-free implicit GEMM: A and B fragments load
//         directly from L2-resident global; 12 v_wmma_f32_16x16x32_bf16
//         per K-step per wave.

typedef __attribute__((ext_vector_type(16))) __bf16 v16bf;
typedef __attribute__((ext_vector_type(8)))  float  v8f;

constexpr int CIN  = 128;
constexpr int COUT = 256;
constexpr int HDIM = 56;
constexpr int WDIM = 56;
constexpr int HW   = HDIM * WDIM;          // 3136
constexpr int NBATCH = 32;
constexpr int NPIX = NBATCH * HW;          // 100352
constexpr int KG   = 9 * CIN;              // 1152, K index = pos*128 + ci
constexpr int PH   = 58;                   // padded height
constexpr int PW   = 58;                   // padded width
constexpr int BM   = 128;                  // C_out block tile
constexpr int BN   = 64;                   // pixel block tile
constexpr int BK   = 32;                   // K step = one WMMA K

constexpr size_t WELEMS  = (size_t)COUT * KG;             // 294912
constexpr size_t XPELEMS = (size_t)NBATCH * PH * PW * CIN;// 13778944 (padded NHWC)
constexpr size_t WHI_OFF = 0;
constexpr size_t WLO_OFF = WELEMS * 2;                    // 589824
constexpr size_t XHI_OFF = WLO_OFF + WELEMS * 2;          // 1179648
constexpr size_t XLO_OFF = XHI_OFF + XPELEMS * 2;         // + 27557888

union Frag { float4 q[2]; v16bf v; };

// ---- pass 1a: weights (256,128,3,3) fp32 -> hi/lo bf16, K reordered to (pos, ci)
__global__ __launch_bounds__(256)
void prep_w(const float* __restrict__ W, __bf16* __restrict__ whi, __bf16* __restrict__ wlo)
{
    int i = blockIdx.x * 256 + threadIdx.x;          // source-contiguous
    if (i >= (int)WELEMS) return;
    int co = i / KG;
    int rr = i - co * KG;
    int ci = rr / 9;
    int pos = rr - ci * 9;
    float v = W[i];
    __bf16 h = (__bf16)v;
    int d = co * KG + pos * CIN + ci;
    whi[d] = h;
    wlo[d] = (__bf16)(v - (float)h);
}

// ---- pass 1b: x (N,C,56,56) fp32 -> hi/lo bf16, zero-padded (N,58,58,C) NHWC
__global__ __launch_bounds__(256)
void prep_x_pad(const float* __restrict__ X, __bf16* __restrict__ xhi, __bf16* __restrict__ xlo)
{
    size_t i = ((size_t)blockIdx.x * 256 + threadIdx.x) * 4;   // XPELEMS % 1024 == 0
    int    ci  = (int)(i & (CIN - 1));      // 4-aligned
    size_t pix = i >> 7;                    // / CIN
    int    wp  = (int)(pix % PW);
    size_t r1  = pix / PW;
    int    hp  = (int)(r1 % PH);
    int    n   = (int)(r1 / PH);
    int h = hp - 1, w = wp - 1;
    bool valid = ((unsigned)h < (unsigned)HDIM) && ((unsigned)w < (unsigned)WDIM);
    int hs = valid ? h : 0;                 // clamp -> loads always in-bounds
    int ws = valid ? w : 0;
    size_t sbase = (((size_t)n * CIN + ci) * HDIM + hs) * WDIM + ws;
    union { __bf16 b[4]; uint2 u; } hh, ll;
    #pragma unroll
    for (int e = 0; e < 4; ++e) {
        float v = X[sbase + (size_t)e * HW];
        v = valid ? v : 0.0f;
        __bf16 hv = (__bf16)v;
        hh.b[e] = hv;
        ll.b[e] = (__bf16)(v - (float)hv);
    }
    *(uint2*)(xhi + i) = hh.u;
    *(uint2*)(xlo + i) = ll.u;
}

// ---- pass 2: LDS-free implicit GEMM
__global__ __launch_bounds__(256)
void conv3x3_wmma_bf16x3(const __bf16* __restrict__ xhi, const __bf16* __restrict__ xlo,
                         const __bf16* __restrict__ whi, const __bf16* __restrict__ wlo,
                         float* __restrict__ Out)
{
    const int t    = threadIdx.x;
    const int lane = t & 31;
    const int wid  = t >> 5;         // 0..7
    const int wm   = wid & 3;        // 32-row C_out sub-tile
    const int wn   = wid >> 2;       // 32-col pixel sub-tile
    const int co0  = blockIdx.y * BM;
    const int px0  = blockIdx.x * BN;
    const int lm   = lane & 15;
    const int lh   = lane >> 4;

    // ---- A fragment base pointers (per lane; includes lh*8 K-group offset)
    const size_t arow = (size_t)(co0 + wm * 32 + lm) * KG;
    const __bf16* __restrict__ wh0 = whi + arow + lh * 8;
    const __bf16* __restrict__ wh1 = wh0 + (size_t)16 * KG;
    const __bf16* __restrict__ wl0 = wlo + arow + lh * 8;
    const __bf16* __restrict__ wl1 = wl0 + (size_t)16 * KG;

    // ---- B pixel base offsets (padded NHWC; per lane, two N-tiles)
    const int p0 = px0 + wn * 32 + lm;
    const int p1 = p0 + 16;
    const int n0 = p0 / HW, hw0 = p0 - n0 * HW, h0 = hw0 / WDIM, w0 = hw0 - h0 * WDIM;
    const int n1 = p1 / HW, hw1 = p1 - n1 * HW, h1 = hw1 / WDIM, w1 = hw1 - h1 * WDIM;
    // pixel (h,w): pos (kh,kw) reads padded coords (h+kh, w+kw)
    const size_t pb0 = (((size_t)n0 * PH + h0) * PW + w0) * CIN + lh * 8;
    const size_t pb1 = (((size_t)n1 * PH + h1) * PW + w1) * CIN + lh * 8;

    v8f acc00 = {}, acc01 = {}, acc10 = {}, acc11 = {};

    for (int pos = 0; pos < 9; ++pos) {
        const int kh = pos / 3;
        const int kw = pos - kh * 3;
        const int posoff = (kh * PW + kw) * CIN;

        #pragma unroll
        for (int ci0 = 0; ci0 < CIN; ci0 += BK) {
            const int k0 = pos * CIN + ci0;

            Frag a0h, a0l, a1h, a1l;
            a0h.q[0] = *(const float4*)(wh0 + k0);
            a0h.q[1] = *(const float4*)(wh0 + k0 + 16);
            a0l.q[0] = *(const float4*)(wl0 + k0);
            a0l.q[1] = *(const float4*)(wl0 + k0 + 16);
            a1h.q[0] = *(const float4*)(wh1 + k0);
            a1h.q[1] = *(const float4*)(wh1 + k0 + 16);
            a1l.q[0] = *(const float4*)(wl1 + k0);
            a1l.q[1] = *(const float4*)(wl1 + k0 + 16);

            const __bf16* __restrict__ bh0 = xhi + pb0 + posoff + ci0;
            const __bf16* __restrict__ bl0 = xlo + pb0 + posoff + ci0;
            const __bf16* __restrict__ bh1 = xhi + pb1 + posoff + ci0;
            const __bf16* __restrict__ bl1 = xlo + pb1 + posoff + ci0;
            Frag b0h, b0l, b1h, b1l;
            b0h.q[0] = *(const float4*)(bh0);
            b0h.q[1] = *(const float4*)(bh0 + 16);
            b0l.q[0] = *(const float4*)(bl0);
            b0l.q[1] = *(const float4*)(bl0 + 16);
            b1h.q[0] = *(const float4*)(bh1);
            b1h.q[1] = *(const float4*)(bh1 + 16);
            b1l.q[0] = *(const float4*)(bl1);
            b1l.q[1] = *(const float4*)(bl1 + 16);

            // bf16x3: hi*hi + hi*lo + lo*hi, fp32 accumulate (12 WMMAs)
            acc00 = __builtin_amdgcn_wmma_f32_16x16x32_bf16(false, a0h.v, false, b0h.v, (short)0, acc00, false, false);
            acc00 = __builtin_amdgcn_wmma_f32_16x16x32_bf16(false, a0h.v, false, b0l.v, (short)0, acc00, false, false);
            acc00 = __builtin_amdgcn_wmma_f32_16x16x32_bf16(false, a0l.v, false, b0h.v, (short)0, acc00, false, false);

            acc01 = __builtin_amdgcn_wmma_f32_16x16x32_bf16(false, a0h.v, false, b1h.v, (short)0, acc01, false, false);
            acc01 = __builtin_amdgcn_wmma_f32_16x16x32_bf16(false, a0h.v, false, b1l.v, (short)0, acc01, false, false);
            acc01 = __builtin_amdgcn_wmma_f32_16x16x32_bf16(false, a0l.v, false, b1h.v, (short)0, acc01, false, false);

            acc10 = __builtin_amdgcn_wmma_f32_16x16x32_bf16(false, a1h.v, false, b0h.v, (short)0, acc10, false, false);
            acc10 = __builtin_amdgcn_wmma_f32_16x16x32_bf16(false, a1h.v, false, b0l.v, (short)0, acc10, false, false);
            acc10 = __builtin_amdgcn_wmma_f32_16x16x32_bf16(false, a1l.v, false, b0h.v, (short)0, acc10, false, false);

            acc11 = __builtin_amdgcn_wmma_f32_16x16x32_bf16(false, a1h.v, false, b1h.v, (short)0, acc11, false, false);
            acc11 = __builtin_amdgcn_wmma_f32_16x16x32_bf16(false, a1h.v, false, b1l.v, (short)0, acc11, false, false);
            acc11 = __builtin_amdgcn_wmma_f32_16x16x32_bf16(false, a1l.v, false, b1h.v, (short)0, acc11, false, false);
        }
    }

    // ---- epilogue: C/D layout: VGPR v -> M = v + 8*lh, N = lane%16
    float* __restrict__ o0 = Out + (size_t)n0 * COUT * HW + hw0;
    float* __restrict__ o1 = Out + (size_t)n1 * COUT * HW + hw1;
    const int cb0 = co0 + wm * 32 + lh * 8;   // M-frag 0 rows
    const int cb1 = cb0 + 16;                 // M-frag 1 rows
    #pragma unroll
    for (int v = 0; v < 8; ++v) {
        o0[(size_t)(cb0 + v) * HW] = acc00[v];
        o1[(size_t)(cb0 + v) * HW] = acc01[v];
        o0[(size_t)(cb1 + v) * HW] = acc10[v];
        o1[(size_t)(cb1 + v) * HW] = acc11[v];
    }
}

extern "C" void kernel_launch(void* const* d_in, const int* in_sizes, int n_in,
                              void* d_out, int out_size, void* d_ws, size_t ws_size,
                              hipStream_t stream) {
    const float* x  = (const float*)d_in[0];   // (32,128,56,56) fp32
    const float* wt = (const float*)d_in[1];   // (256,128,3,3)  fp32
    float* out = (float*)d_out;                // (32,256,56,56) fp32

    char* ws = (char*)d_ws;
    __bf16* whi = (__bf16*)(ws + WHI_OFF);
    __bf16* wlo = (__bf16*)(ws + WLO_OFF);
    __bf16* xhi = (__bf16*)(ws + XHI_OFF);
    __bf16* xlo = (__bf16*)(ws + XLO_OFF);

    prep_w<<<dim3((unsigned)((WELEMS + 255) / 256)), 256, 0, stream>>>(wt, whi, wlo);
    prep_x_pad<<<dim3((unsigned)(XPELEMS / 4 / 256)), 256, 0, stream>>>(x, xhi, xlo);

    dim3 grid(NPIX / BN, COUT / BM);           // (1568, 2)
    conv3x3_wmma_bf16x3<<<grid, 256, 0, stream>>>(xhi, xlo, whi, wlo, out);
}
